// CorrNet_73057393705037
// MI455X (gfx1250) — matrix operand order repaired and checked by reading, compiled, not run on hardware
//
#include <hip/hip_runtime.h>
#include <hip/hip_bf16.h>

typedef __attribute__((ext_vector_type(16))) __bf16 v16bf;
typedef __attribute__((ext_vector_type(8)))  float  v8f;
typedef __attribute__((ext_vector_type(2)))  float  v2f;
typedef __attribute__((ext_vector_type(4)))  unsigned int u32x4;
typedef __attribute__((ext_vector_type(8)))  int    i32x8;
typedef __attribute__((ext_vector_type(4)))  int    i32x4;

static constexpr int NPTS = 16384;   // N == M == 16384
static constexpr int TILE = 64;      // points per block in MLP kernels

__device__ inline __bf16 to_bf16(float x) { return (__bf16)x; }

// ---------------------------------------------------------------------------
// GroupNorm over 32-channel groups, activations resident in LDS [64 x C].
// Caller handles barriers.
// ---------------------------------------------------------------------------
__device__ inline void group_norm_lds(float* act, int C, const float* gamma,
                                      const float* beta) {
  const int ngrp = C >> 5;
  for (int rg = threadIdx.x; rg < TILE * ngrp; rg += blockDim.x) {
    const int row = rg / ngrp, grp = rg % ngrp;
    float* p = act + row * C + grp * 32;
    float mu = 0.f;
    #pragma unroll 8
    for (int j = 0; j < 32; ++j) mu += p[j];
    mu *= (1.f / 32.f);
    float var = 0.f;
    #pragma unroll 8
    for (int j = 0; j < 32; ++j) { float d = p[j] - mu; var += d * d; }
    var *= (1.f / 32.f);
    const float inv = rsqrtf(var + 1e-5f);
    const float* gg = gamma + grp * 32;
    const float* bb = beta + grp * 32;
    #pragma unroll 8
    for (int j = 0; j < 32; ++j) p[j] = (p[j] - mu) * inv * gg[j] + bb[j];
  }
}

// ---------------------------------------------------------------------------
// Generic dense layer via v_wmma_f32_16x16x32_bf16.
//   aIn : LDS [64 x Cin] f32 activations
//   W   : global [Cout x Cin] f32 (row-major, as in reference: out = x @ W.T)
//   aOut: LDS [64 x Cout]
// 8 waves/block, tiles are 16x16; A/B fragments packed per CDNA5 ISA layout.
// ---------------------------------------------------------------------------
__device__ inline void wmma_layer(const float* aIn, int Cin, const float* W,
                                  const float* bias, float* aOut, int Cout,
                                  bool relu) {
  const int lane = threadIdx.x & 31;
  const int wave = threadIdx.x >> 5;
  const int lrow = lane & 15;   // lane % 16
  const int lhi  = lane >> 4;   // lane / 16
  const int ntiles = (TILE / 16) * (Cout >> 4);
  for (int t = wave; t < ntiles; t += 8) {
    const int rt = t & 3;    // row tile (16 points)
    const int ct = t >> 2;   // col tile (16 out channels)
    const int m  = rt * 16 + lrow;
    const int n  = ct * 16 + lrow;
    v8f c = {};
    for (int kc = 0; kc < Cin; kc += 32) {
      v16bf a, b;
      #pragma unroll
      for (int v = 0; v < 8; ++v) {
        // A 16x32 bf16 layout: VGPR v holds K pair {K0,K0+1},
        // K0 = 2v (+8 if v>=4) + 8*(lane/16)
        const int ka = kc + 2 * v + (v >= 4 ? 8 : 0) + 8 * lhi;
        a[2 * v]     = to_bf16(aIn[m * Cin + ka]);
        a[2 * v + 1] = to_bf16(aIn[m * Cin + ka + 1]);
        // B 32x16 bf16 layout: K0 = 2v (+8 if v>=4) + 8*(lane/16); B[k][n] = W[n][k]
        const int kb = kc + 2 * v + (v >= 4 ? 8 : 0) + 8 * lhi;
        b[2 * v]     = to_bf16(W[n * Cin + kb]);
        b[2 * v + 1] = to_bf16(W[n * Cin + kb + 1]);
      }
      c = __builtin_amdgcn_wmma_f32_16x16x32_bf16(false, a, false, b,
                                                  (short)0, c, false, false);
    }
    #pragma unroll
    for (int v = 0; v < 8; ++v) {
      const int row = rt * 16 + v + 8 * lhi;   // C/D layout: M = v + 8*(lane/16)
      const int col = ct * 16 + lrow;
      float x = c[v] + bias[col];
      if (relu) x = x > 0.f ? x : 0.2f * x;
      aOut[row * Cout + col] = x;
    }
  }
}

// ---------------------------------------------------------------------------
// Head: layers 1-4 (3->32->64->128, 128->128), emit f and per-block channel max.
// ---------------------------------------------------------------------------
__global__ void head_kernel(const float* __restrict__ xyz,
                            const float* W1, const float* b1, const float* g1, const float* be1,
                            const float* W2, const float* b2, const float* g2, const float* be2,
                            const float* W3, const float* b3, const float* g3, const float* be3,
                            const float* W4, const float* b4, const float* g4, const float* be4,
                            float* __restrict__ f_out, float* __restrict__ bmax) {
  extern __shared__ float sm[];
  float* A = sm;                 // 64*128
  float* B = sm + TILE * 128;    // 64*128
  const int p0 = blockIdx.x * TILE;

  // layer1: 3 -> 32 (VALU; K=3 too small for matrix cores)
  for (int e = threadIdx.x; e < TILE * 32; e += blockDim.x) {
    const int r = e >> 5, c = e & 31;
    const float* x = xyz + (size_t)(p0 + r) * 3;
    float s = b1[c] + x[0] * W1[c * 3] + x[1] * W1[c * 3 + 1] + x[2] * W1[c * 3 + 2];
    A[r * 32 + c] = s > 0.f ? s : 0.2f * s;
  }
  __syncthreads();
  group_norm_lds(A, 32, g1, be1);
  __syncthreads();
  wmma_layer(A, 32, W2, b2, B, 64, true);
  __syncthreads();
  group_norm_lds(B, 64, g2, be2);
  __syncthreads();
  wmma_layer(B, 64, W3, b3, A, 128, true);
  __syncthreads();
  group_norm_lds(A, 128, g3, be3);
  __syncthreads();
  // f features out
  for (int e = threadIdx.x; e < TILE * 128; e += blockDim.x)
    f_out[(size_t)p0 * 128 + e] = A[e];
  wmma_layer(A, 128, W4, b4, B, 128, true);
  __syncthreads();
  group_norm_lds(B, 128, g4, be4);
  __syncthreads();
  // block-local channel max of h
  for (int c = threadIdx.x; c < 128; c += blockDim.x) {
    float mx = -3.4e38f;
    for (int r = 0; r < TILE; ++r) mx = fmaxf(mx, B[r * 128 + c]);
    bmax[blockIdx.x * 128 + c] = mx;
  }
}

__global__ void reduce_max_kernel(const float* __restrict__ bmax, int nblk,
                                  float* __restrict__ g) {
  const int c = threadIdx.x;  // 128 threads
  float mx = -3.4e38f;
  for (int b = 0; b < nblk; ++b) mx = fmaxf(mx, bmax[b * 128 + c]);
  g[c] = mx;
}

// ---------------------------------------------------------------------------
// Tail: concat [f, g] -> 256->128->64->128, L2-normalize rows, write unit feats.
// ---------------------------------------------------------------------------
__global__ void tail_kernel(const float* __restrict__ f, const float* __restrict__ g,
                            const float* W5, const float* b5, const float* g5, const float* be5,
                            const float* W6, const float* b6, const float* g6, const float* be6,
                            const float* W7, const float* b7,
                            float* __restrict__ out) {
  extern __shared__ float sm[];
  float* A = sm;                 // 64*256
  float* B = sm + TILE * 256;    // 64*128
  const int p0 = blockIdx.x * TILE;
  for (int e = threadIdx.x; e < TILE * 256; e += blockDim.x) {
    const int r = e >> 8, c = e & 255;
    A[e] = (c < 128) ? f[(size_t)(p0 + r) * 128 + c] : g[c - 128];
  }
  __syncthreads();
  wmma_layer(A, 256, W5, b5, B, 128, true);
  __syncthreads();
  group_norm_lds(B, 128, g5, be5);
  __syncthreads();
  wmma_layer(B, 128, W6, b6, A, 64, true);
  __syncthreads();
  group_norm_lds(A, 64, g6, be6);
  __syncthreads();
  wmma_layer(A, 64, W7, b7, B, 128, false);   // final projection, no act/GN
  __syncthreads();
  for (int r = threadIdx.x; r < TILE; r += blockDim.x) {
    float s = 0.f;
    #pragma unroll 4
    for (int c = 0; c < 128; ++c) { const float x = B[r * 128 + c]; s += x * x; }
    const float inv = rsqrtf(s);
    #pragma unroll 4
    for (int c = 0; c < 128; ++c)
      out[(size_t)(p0 + r) * 128 + c] = B[r * 128 + c] * inv;
  }
}

// ---------------------------------------------------------------------------
// Fused cosine-similarity GEMM + row argmax using f32 WMMA (16x16x4).
// Block = 8 waves; wave w owns rows [blk*128 + 16w, +16). A fragments (16 rows
// x K=128) live in 64 VGPRs. op tiles (128 rows x 128 ch = 64 KB) are DMA'd
// into LDS by the Tensor Data Mover, double-buffered and tracked by TENSORcnt
// so the next tile's DMA overlaps the current tile's 256 WMMAs.
// ---------------------------------------------------------------------------
__global__ void sim_argmax_kernel(const float* __restrict__ ov,
                                  const float* __restrict__ op, int M,
                                  float* __restrict__ S, int* __restrict__ idx) {
  extern __shared__ float sm[];   // 2 x (128*128) op tiles
  const int lane = threadIdx.x & 31;
  const int wave = threadIdx.x >> 5;
  const int lrow = lane & 15;
  const int lhi  = lane >> 4;
  const int r0 = blockIdx.x * 128 + wave * 16;
  const int nIter = M >> 7;

  // A 16x4 f32 layout per step: lane holds M=lane%16, K = 2*(lane/16)+{0,1}
  v2f aF[32];
  {
    const float* base = ov + (size_t)(r0 + lrow) * 128 + 2 * lhi;
    #pragma unroll
    for (int j = 0; j < 32; ++j) aF[j] = *(const v2f*)(base + 4 * j);
  }

  float best[8];
  int   bidx[8];
  #pragma unroll
  for (int v = 0; v < 8; ++v) { best[v] = -3.4e38f; bidx[v] = 0; }

#if __has_builtin(__builtin_amdgcn_tensor_load_to_lds)
  // --- TDM path: one 64 KB 1-D tile per descriptor, issued by wave 0 only ---
  const unsigned ldsBase = (unsigned)(unsigned long long)(size_t)sm;  // LDS aperture: low 32 bits = LDS byte addr
  auto issue_tile = [&](int it) {
    const unsigned long long ga =
        (unsigned long long)(size_t)(op + (size_t)it * 128 * 128);
    u32x4 g0;
    g0[0] = 1u;                                            // count=1, user mode
    g0[1] = ldsBase + (unsigned)((it & 1) * 128 * 128 * 4);// lds_addr (bytes)
    g0[2] = (unsigned)ga;                                  // global_addr[31:0]
    g0[3] = (unsigned)((ga >> 32) & 0x1FFFFFFu) | (2u << 30); // addr[56:32] | type=2
    i32x8 g1;
    g1[0] = (int)(2u << 16);        // data_size = 4B
    g1[1] = (int)(16384u << 16);    // tensor_dim0[15:0] = 16384 elements
    g1[2] = (int)(1u << 16);        // tensor_dim0 hi = 0, tensor_dim1 = 1
    g1[3] = (int)(16384u << 16);    // tile_dim0 = 16384 (one linear 64 KB tile)
    g1[4] = 0;                      // tile_dim1/tile_dim2 unused
    g1[5] = 16384;                  // tensor_dim0_stride
    g1[6] = 0;
    g1[7] = 0;
    i32x4 z4 = {0, 0, 0, 0};
    i32x8 z8 = {0, 0, 0, 0, 0, 0, 0, 0};
    __builtin_amdgcn_tensor_load_to_lds(g0, g1, z4, z4, z8, 0);  // 6-arg form
  };
  if (wave == 0) issue_tile(0);
#endif

  for (int it = 0; it < nIter; ++it) {
#if __has_builtin(__builtin_amdgcn_tensor_load_to_lds)
    if (wave == 0) {
      if (it + 1 < nIter) {
        issue_tile(it + 1);                       // prefetch next tile
        __builtin_amdgcn_s_wait_tensorcnt(1);     // current tile done, next in flight
      } else {
        __builtin_amdgcn_s_wait_tensorcnt(0);
      }
    }
    __syncthreads();
    const float* tile = sm + (it & 1) * (128 * 128);
#else
    // --- fallback: cooperative vector loads ---
    __syncthreads();
    for (int e = threadIdx.x * 4; e < 128 * 128; e += blockDim.x * 4)
      *(float4*)(sm + e) = *(const float4*)(op + (size_t)it * 128 * 128 + e);
    __syncthreads();
    const float* tile = sm;
#endif

    const int ct = it << 7;
    for (int st = 0; st < 8; ++st) {
      const int c0 = st * 16;
      v8f c = {};
      // B 4x16 f32 layout: lane holds N=lane%16, K = v + 2*(lane/16)
      const float* bbase = tile + (c0 + lrow) * 128 + 2 * lhi;
      #pragma unroll
      for (int j = 0; j < 32; ++j) {
        const v2f b = *(const v2f*)(bbase + 4 * j);
        c = __builtin_amdgcn_wmma_f32_16x16x4_f32(false, aF[j], false, b,
                                                  (short)0, c, false, false);
      }
      // per-row argmax across the 16 columns (lanes) of this tile
      #pragma unroll
      for (int v = 0; v < 8; ++v) {
        float val = c[v];
        int   col = ct + c0 + lrow;
        #pragma unroll
        for (int off = 1; off < 16; off <<= 1) {
          const float v2 = __shfl_xor(val, off, 32);
          const int   c2 = __shfl_xor(col, off, 32);
          if (v2 > val || (v2 == val && c2 < col)) { val = v2; col = c2; }
        }
        if (val > best[v] || (val == best[v] && col < bidx[v])) {
          best[v] = val; bidx[v] = col;
        }
      }
    }
    __syncthreads();   // all waves done with this buffer before it is re-DMA'd
  }

  if (lrow == 0) {
    #pragma unroll
    for (int v = 0; v < 8; ++v) {
      const int row = r0 + v + 8 * lhi;   // C layout: M = v + 8*(lane/16)
      S[row]   = best[v];
      idx[row] = bidx[v];
    }
  }
}

// ---------------------------------------------------------------------------
// Final scorer: [ov | op[idx] | S] (257) -> 64 (leaky+GN groups=2) -> 1.
// One block per row, 64 threads; each wave32 is exactly one GN group.
// ---------------------------------------------------------------------------
__global__ void final_kernel(const float* __restrict__ ov, const float* __restrict__ op,
                             const float* __restrict__ S, const int* __restrict__ idx,
                             const float* W8, const float* b8, const float* g8,
                             const float* be8, const float* W9, const float* b9,
                             float* __restrict__ oc) {
  __shared__ float red[2];
  const int row = blockIdx.x;
  const int t = threadIdx.x;   // 64
  const float* y = ov + (size_t)row * 128;
  const float* x = op + (size_t)idx[row] * 128;
  const float* w = W8 + t * 257;
  float s = b8[t];
  #pragma unroll 4
  for (int j = 0; j < 128; ++j) s += y[j] * w[j];
  #pragma unroll 4
  for (int j = 0; j < 128; ++j) s += x[j] * w[128 + j];
  s += S[row] * w[256];
  s = s > 0.f ? s : 0.2f * s;
  // GroupNorm(groups=2): wave-wide mean/var via shuffles
  float mu = s;
  #pragma unroll
  for (int off = 1; off < 32; off <<= 1) mu += __shfl_xor(mu, off, 32);
  mu *= (1.f / 32.f);
  const float d = s - mu;
  float var = d * d;
  #pragma unroll
  for (int off = 1; off < 32; off <<= 1) var += __shfl_xor(var, off, 32);
  var *= (1.f / 32.f);
  const float h = d * rsqrtf(var + 1e-5f) * g8[t] + be8[t];
  float part = h * W9[t];
  #pragma unroll
  for (int off = 1; off < 32; off <<= 1) part += __shfl_xor(part, off, 32);
  if ((t & 31) == 0) red[t >> 5] = part;
  __syncthreads();
  if (t == 0) oc[row] = red[0] + red[1] + b9[0];
}

// ---------------------------------------------------------------------------
extern "C" void kernel_launch(void* const* d_in, const int* in_sizes, int n_in,
                              void* d_out, int out_size, void* d_ws, size_t ws_size,
                              hipStream_t stream) {
  const float* vtx = (const float*)d_in[0];
  const float* pts = (const float*)d_in[1];
  const float *W1 = (const float*)d_in[2],  *b1 = (const float*)d_in[3],
              *g1 = (const float*)d_in[4],  *be1 = (const float*)d_in[5];
  const float *W2 = (const float*)d_in[6],  *b2 = (const float*)d_in[7],
              *g2 = (const float*)d_in[8],  *be2 = (const float*)d_in[9];
  const float *W3 = (const float*)d_in[10], *b3 = (const float*)d_in[11],
              *g3 = (const float*)d_in[12], *be3 = (const float*)d_in[13];
  const float *W4 = (const float*)d_in[14], *b4 = (const float*)d_in[15],
              *g4 = (const float*)d_in[16], *be4 = (const float*)d_in[17];
  const float *W5 = (const float*)d_in[18], *b5 = (const float*)d_in[19],
              *g5 = (const float*)d_in[20], *be5 = (const float*)d_in[21];
  const float *W6 = (const float*)d_in[22], *b6 = (const float*)d_in[23],
              *g6 = (const float*)d_in[24], *be6 = (const float*)d_in[25];
  const float *W7 = (const float*)d_in[26], *b7 = (const float*)d_in[27];
  const float *W8 = (const float*)d_in[28], *b8 = (const float*)d_in[29],
              *g8 = (const float*)d_in[30], *be8 = (const float*)d_in[31];
  const float *W9 = (const float*)d_in[32], *b9 = (const float*)d_in[33];

  const int N = NPTS, M = NPTS;
  float* ov = (float*)d_out;
  float* op = ov + (size_t)N * 128;
  float* oc = op + (size_t)M * 128;

  float* f_v  = (float*)d_ws;
  float* f_p  = f_v + (size_t)N * 128;
  float* bm_v = f_p + (size_t)M * 128;
  float* bm_p = bm_v + (size_t)(N / TILE) * 128;
  float* g_v  = bm_p + (size_t)(M / TILE) * 128;
  float* g_p  = g_v + 128;
  float* Sb   = g_p + 128;
  int*   ixb  = (int*)(Sb + N);

  const dim3 blk(256);
  const size_t lds_head = (size_t)(TILE * 128 * 2) * sizeof(float);          // 64 KB
  const size_t lds_tail = (size_t)(TILE * 256 + TILE * 128) * sizeof(float); // 96 KB
  const size_t lds_sim  = (size_t)(2 * 128 * 128) * sizeof(float);           // 128 KB (double buffer)

  head_kernel<<<N / TILE, blk, lds_head, stream>>>(vtx,
      W1, b1, g1, be1, W2, b2, g2, be2, W3, b3, g3, be3, W4, b4, g4, be4,
      f_v, bm_v);
  head_kernel<<<M / TILE, blk, lds_head, stream>>>(pts,
      W1, b1, g1, be1, W2, b2, g2, be2, W3, b3, g3, be3, W4, b4, g4, be4,
      f_p, bm_p);
  reduce_max_kernel<<<1, 128, 0, stream>>>(bm_v, N / TILE, g_v);
  reduce_max_kernel<<<1, 128, 0, stream>>>(bm_p, M / TILE, g_p);
  tail_kernel<<<N / TILE, blk, lds_tail, stream>>>(f_v, g_v,
      W5, b5, g5, be5, W6, b6, g6, be6, W7, b7, ov);
  tail_kernel<<<M / TILE, blk, lds_tail, stream>>>(f_p, g_p,
      W5, b5, g5, be5, W6, b6, g6, be6, W7, b7, op);
  sim_argmax_kernel<<<N / 128, blk, lds_sim, stream>>>(ov, op, M, Sb, ixb);
  final_kernel<<<N, 64, 0, stream>>>(ov, op, Sb, ixb, W8, b8, g8, be8, W9, b9, oc);
}
